// NeuronMemory_12979391168721
// MI455X (gfx1250) — matrix operand order repaired and checked by reading, not compiled, not run on hardware
//
#include <hip/hip_runtime.h>
#include <hip/hip_bf16.h>

// Problem constants (from reference):
#define D_DIM   1024
#define NKTOT   65536
#define RANK    128
#define CK      20      // COARSE_K
#define FK      10      // FINE_K
#define BTOK    4096    // B*S = 2*2048
#define NSPLIT  8       // N-splits of the routing GEMM per token group
#define NPS     (NKTOT / NSPLIT)   // 8192
#define CHUNK_N 256
#define BM      64
#define CAP     32      // per-row candidate buffer capacity per chunk

typedef __attribute__((ext_vector_type(16))) __bf16 v16bf;
typedef __attribute__((ext_vector_type(8)))  float  v8f;

union FragU { uint4 q[2]; v16bf v; };

__device__ __forceinline__ unsigned short f2bf(float f) {
  unsigned int u = __builtin_bit_cast(unsigned int, f);
  u += 0x7FFFu + ((u >> 16) & 1u);   // round-to-nearest-even
  return (unsigned short)(u >> 16);
}

// Insert (v,id) into sorted-desc top-CK list with JAX tie rule (value desc, index asc).
__device__ __forceinline__ void topk_insert(float* tv, int* ti, float v, int id) {
  if (v > tv[CK-1] || (v == tv[CK-1] && id < ti[CK-1])) {
    int p = CK - 1;
    while (p > 0 && (v > tv[p-1] || (v == tv[p-1] && id < ti[p-1]))) {
      tv[p] = tv[p-1]; ti[p] = ti[p-1]; --p;
    }
    tv[p] = v; ti[p] = id;
  }
}

// ---------------------------------------------------------------------------
// Kernel 1a: W_router [D, NK] fp32  ->  bf16 in WMMA B-fragment order.
// Fragment (n_tile = n/16, k_tile = k/32): lane L holds 16 contiguous bf16:
//   n = n_tile*16 + (L&15), K = k_tile*32 + (L>>4)*16 + 0..15
// stored at [(n_tile*32 + k_tile)*512 + L*16].
// ---------------------------------------------------------------------------
__global__ void k_convert_w(const float* __restrict__ W, unsigned int* __restrict__ out) {
  long g  = (long)blockIdx.x * 256 + threadIdx.x;   // dword id (2 bf16)
  long eb = g * 2;
  int  wb = (int)(eb & 511);
  long frag  = eb >> 9;
  int  kt    = (int)(frag & 31);
  long ntile = frag >> 5;
  int  lane  = wb >> 4;
  int  e     = wb & 15;
  long k = (long)kt * 32 + (lane >> 4) * 16 + e;
  long n = ntile * 16 + (lane & 15);
  float f0 = W[k * NKTOT + n];
  float f1 = W[(k + 1) * NKTOT + n];
  out[g] = (unsigned int)f2bf(f0) | ((unsigned int)f2bf(f1) << 16);
}

// Kernel 1b: x fp32 -> bf16, row-major [token][D]
__global__ void k_convert_x(const float* __restrict__ x, unsigned int* __restrict__ out) {
  int g = blockIdx.x * 256 + threadIdx.x;
  float2 v = ((const float2*)x)[g];
  out[g] = (unsigned int)f2bf(v.x) | ((unsigned int)f2bf(v.y) << 16);
}

// ---------------------------------------------------------------------------
// Kernel 2: fused routing GEMM (bf16 WMMA) + per-token running top-20.
// grid = (64 token groups, 8 N-splits), 256 threads = 8 waves, waves 2(M)x4(N).
// Each wave: TM=2 x TN=4 WMMA tiles (32M x 64N), so 8 WMMAs share 2 A-frags.
// ---------------------------------------------------------------------------
#define AS_STRIDE 1032   // 1024+8 bf16 pad -> A-frag ds_read_b128 conflict-free
#define LT_STRIDE 260    // mod 64 = 4 -> scan bank = 4*row + q, conflict-free
#define OFF_LT    (BM*AS_STRIDE*2)                   // 132096
#define OFF_TV    (OFF_LT + BM*LT_STRIDE*4)          // 198656
#define OFF_TI    (OFF_TV + BM*CK*4)                 // 203776
#define OFF_CV    (OFF_TI + BM*CK*4)                 // 208896
#define OFF_CI    (OFF_CV + BM*CAP*4)                // 217088
#define OFF_CNT   (OFF_CI + BM*CAP*4)                // 225280
#define SMEM_BYTES (OFF_CNT + BM*4)                  // 225536

__global__ __launch_bounds__(256) void k_route_topk(
    const unsigned short* __restrict__ Wtf, const unsigned short* __restrict__ xb,
    float* __restrict__ pv, int* __restrict__ pi) {
  extern __shared__ char smem[];
  unsigned short* As = (unsigned short*)smem;
  float* ltile = (float*)(smem + OFF_LT);
  float* topv  = (float*)(smem + OFF_TV);
  int*   topi  = (int*)  (smem + OFF_TI);
  float* candv = (float*)(smem + OFF_CV);
  int*   candi = (int*)  (smem + OFF_CI);
  int*   cnt   = (int*)  (smem + OFF_CNT);

  const int tid   = threadIdx.x;
  const int mg    = blockIdx.x;
  const int split = blockIdx.y;
  const int Mbase = mg * BM;

  // Stage x-tile (64 x 1024 bf16) into LDS.
  for (int idx = tid; idx < BM * 128; idx += 256) {
    int r = idx >> 7, c = idx & 127;
    *(uint4*)&As[r * AS_STRIDE + c * 8] = *(const uint4*)&xb[(Mbase + r) * D_DIM + c * 8];
  }
  if (tid < BM) {
    for (int j = 0; j < CK; ++j) { topv[tid*CK + j] = -3.0e38f; topi[tid*CK + j] = 0; }
    cnt[tid] = 0;
  }
  __syncthreads();

  const int lane  = tid & 31;
  const int wave  = tid >> 5;
  const int m_off = (wave >> 2) * 32;    // 0 or 32
  const int n_off = (wave & 3) * 64;     // 0,64,128,192
  const int lrow  = lane & 15;
  const int kbA   = (lane >> 4) * 16;

  const unsigned short* a0base = As + (m_off +      lrow) * AS_STRIDE + kbA;
  const unsigned short* a1base = As + (m_off + 16 + lrow) * AS_STRIDE + kbA;

  for (int chunk = 0; chunk < NPS / CHUNK_N; ++chunk) {
    const int base_n = split * NPS + chunk * CHUNK_N;
    v8f acc0[4] = {{}, {}, {}, {}};   // m tile 0, n sub-tiles 0..3
    v8f acc1[4] = {{}, {}, {}, {}};   // m tile 1

    const long nt0 = (long)(base_n + n_off) >> 4;   // global 16-wide n_tile
    const unsigned short* bbase = Wtf + (nt0 * 32) * 512 + lane * 16;

    #pragma unroll 4
    for (int kt = 0; kt < 32; ++kt) {
      FragU a0, a1;
      a0.q[0] = *(const uint4*)(a0base + kt*32);
      a0.q[1] = *(const uint4*)(a0base + kt*32 + 8);
      a1.q[0] = *(const uint4*)(a1base + kt*32);
      a1.q[1] = *(const uint4*)(a1base + kt*32 + 8);
      #pragma unroll
      for (int j = 0; j < 4; ++j) {
        FragU b;
        b.q[0] = *(const uint4*)(bbase + j*16384 + kt*512);
        b.q[1] = *(const uint4*)(bbase + j*16384 + kt*512 + 8);
        acc0[j] = __builtin_amdgcn_wmma_f32_16x16x32_bf16(false, a0.v, false, b.v, (short)0, acc0[j], false, false);
        acc1[j] = __builtin_amdgcn_wmma_f32_16x16x32_bf16(false, a1.v, false, b.v, (short)0, acc1[j], false, false);
      }
    }

    // Spill 64x256 logits tile to LDS (C layout: M = (lane>>4)*8 + r, N = lane&15)
    const int crow = (lane >> 4) * 8;
    #pragma unroll
    for (int j = 0; j < 4; ++j) {
      #pragma unroll
      for (int r = 0; r < 8; ++r) {
        ltile[(m_off +      crow + r) * LT_STRIDE + (n_off + j*16 + lrow)] = acc0[j][r];
        ltile[(m_off + 16 + crow + r) * LT_STRIDE + (n_off + j*16 + lrow)] = acc1[j][r];
      }
    }
    __syncthreads();

    // Phase 1: all 256 threads filter against the (stale, conservative) 20th-best.
    {
      const int row = tid >> 2, q = tid & 3;
      const float thresh = topv[row*CK + (CK-1)];
      const float* lr = ltile + row * LT_STRIDE;
      for (int i = 0; i < CHUNK_N/4; ++i) {
        int c = q + i*4;                       // interleaved -> conflict-free banks
        float v = lr[c];
        if (v > thresh) {
          int p = atomicAdd(&cnt[row], 1);
          if (p < CAP) { candv[row*CAP + p] = v; candi[row*CAP + p] = base_n + c; }
        }
      }
    }
    __syncthreads();

    // Phase 2: one thread per row inserts survivors (order-independent comparator).
    if (tid < BM) {
      float* tv = topv + tid*CK;
      int*   ti = topi + tid*CK;
      int k = cnt[tid];
      if (k > CAP) {               // overflow (e.g. first chunk): full serial rescan
        const float* lr = ltile + tid * LT_STRIDE;
        for (int c = 0; c < CHUNK_N; ++c) topk_insert(tv, ti, lr[c], base_n + c);
      } else {
        for (int j = 0; j < k; ++j) topk_insert(tv, ti, candv[tid*CAP + j], candi[tid*CAP + j]);
      }
      cnt[tid] = 0;
    }
    __syncthreads();
  }

  if (tid < BM) {
    long token = Mbase + tid;
    for (int j = 0; j < CK; ++j) {
      pv[(token * NSPLIT + split) * CK + j] = topv[tid*CK + j];
      pi[(token * NSPLIT + split) * CK + j] = topi[tid*CK + j];
    }
  }
}

// ---------------------------------------------------------------------------
// Kernel 3: merge 8 partial top-20 lists per token.
// ---------------------------------------------------------------------------
__global__ void k_merge(const float* __restrict__ pv, const int* __restrict__ pi,
                        int* __restrict__ cand) {
  int t = blockIdx.x * 256 + threadIdx.x;
  if (t >= BTOK) return;
  float bv[CK]; int bi[CK];
  #pragma unroll
  for (int j = 0; j < CK; ++j) { bv[j] = -3.0e38f; bi[j] = 0x7fffffff; }
  const float* pvt = pv + (long)t * NSPLIT * CK;
  const int*   pit = pi + (long)t * NSPLIT * CK;
  for (int s = 0; s < NSPLIT * CK; ++s) {
    float v = pvt[s]; int id = pit[s];
    if (v > bv[CK-1] || (v == bv[CK-1] && id < bi[CK-1])) {
      int p = CK - 1;
      while (p > 0 && (v > bv[p-1] || (v == bv[p-1] && id < bi[p-1]))) {
        bv[p] = bv[p-1]; bi[p] = bi[p-1]; --p;
      }
      bv[p] = v; bi[p] = id;
    }
  }
  for (int j = 0; j < CK; ++j) cand[t*CK + j] = bi[j];
}

// ---------------------------------------------------------------------------
// Kernel 4: stage 2 — query, fine scores, top-10 + softmax, weighted V gather.
// ---------------------------------------------------------------------------
__global__ __launch_bounds__(128) void k_stage2(
    const float* __restrict__ x, const float* __restrict__ W_enc,
    const float* __restrict__ K_all, const float* __restrict__ V_all,
    const int* __restrict__ cand, float* __restrict__ out) {
  __shared__ float xs[D_DIM];
  __shared__ float qs[RANK];
  __shared__ int   cs[CK];
  __shared__ float fsc[CK];
  __shared__ float wf_s[FK];
  __shared__ int   gi_s[FK];

  const int  t     = threadIdx.x;
  const long token = blockIdx.x;

  for (int i = t; i < D_DIM/4; i += 128)
    ((float4*)xs)[i] = ((const float4*)(x + token * D_DIM))[i];
  if (t < CK) cs[t] = cand[token * CK + t];
  __syncthreads();

  float q = 0.f;
  #pragma unroll 4
  for (int d = 0; d < D_DIM; ++d) q += xs[d] * W_enc[d * RANK + t];
  qs[t] = q;
  __syncthreads();

  const int lane = t & 31, wv = t >> 5;
  for (int c = wv; c < CK; c += 4) {
    const float* Kr = K_all + (long)cs[c] * RANK;
    float p = 0.f;
    #pragma unroll
    for (int e = 0; e < 4; ++e) { int d = lane + e*32; p += qs[d] * Kr[d]; }
    p += __shfl_xor(p, 16); p += __shfl_xor(p, 8); p += __shfl_xor(p, 4);
    p += __shfl_xor(p, 2);  p += __shfl_xor(p, 1);
    if (lane == 0) fsc[c] = p * 0.08838834764831845f;   // 1/sqrt(128)
  }
  __syncthreads();

  if (t == 0) {
    float sv[FK]; int sl[FK];
    #pragma unroll
    for (int j = 0; j < FK; ++j) { sv[j] = -3.0e38f; sl[j] = 0; }
    for (int c = 0; c < CK; ++c) {
      float v = fsc[c];
      if (v > sv[FK-1]) {
        int p = FK - 1;
        while (p > 0 && v > sv[p-1]) { sv[p] = sv[p-1]; sl[p] = sl[p-1]; --p; }
        sv[p] = v; sl[p] = c;
      }
    }
    float mx = sv[0], sum = 0.f, w[FK];
    #pragma unroll
    for (int j = 0; j < FK; ++j) { w[j] = expf(sv[j] - mx); sum += w[j]; }
    float inv = 1.f / sum;
    #pragma unroll
    for (int j = 0; j < FK; ++j) { wf_s[j] = w[j] * inv; gi_s[j] = cs[sl[j]]; }
  }
  __syncthreads();

  #pragma unroll
  for (int rep = 0; rep < D_DIM/128; ++rep) {
    int d = t + rep * 128;
    float acc = 0.f;
    #pragma unroll
    for (int f = 0; f < FK; ++f) acc += wf_s[f] * V_all[(long)gi_s[f] * D_DIM + d];
    out[token * D_DIM + d] = acc;
  }
}

// ---------------------------------------------------------------------------
// Workspace layout (bytes):
//   [0,          134217728)  W_router bf16, B-fragment order (128 MB, L2-resident)
//   [134217728,  142606336)  x bf16 (8 MB)
//   [142606336,  145227776)  partial top-20 values  f32 [4096][8][20]
//   [145227776,  147849216)  partial top-20 indices i32 [4096][8][20]
//   [147849216,  148176896)  merged candidate_idx   i32 [4096][20]
// ---------------------------------------------------------------------------
extern "C" void kernel_launch(void* const* d_in, const int* in_sizes, int n_in,
                              void* d_out, int out_size, void* d_ws, size_t ws_size,
                              hipStream_t stream) {
  const float* x        = (const float*)d_in[0];
  const float* W_router = (const float*)d_in[1];
  const float* W_enc    = (const float*)d_in[2];
  const float* K_all    = (const float*)d_in[3];
  const float* V_all    = (const float*)d_in[4];
  float* out = (float*)d_out;

  char* ws = (char*)d_ws;
  unsigned short* Wtf = (unsigned short*)ws;
  unsigned short* xb  = (unsigned short*)(ws + 134217728L);
  float* pv  = (float*)(ws + 134217728L + 8388608L);
  int*   pi  = (int*)  (ws + 134217728L + 8388608L + 2621440L);
  int*   cnd = (int*)  (ws + 134217728L + 8388608L + 2L*2621440L);

  (void)hipFuncSetAttribute((const void*)k_route_topk,
                            hipFuncAttributeMaxDynamicSharedMemorySize, SMEM_BYTES);

  k_convert_w<<<dim3(NKTOT/16 * (D_DIM/32) * 512 / 2 / 256), 256, 0, stream>>>(W_router, (unsigned int*)Wtf);
  k_convert_x<<<dim3(BTOK*D_DIM/2/256), 256, 0, stream>>>(x, (unsigned int*)xb);
  k_route_topk<<<dim3(BTOK/BM, NSPLIT), 256, SMEM_BYTES, stream>>>(Wtf, xb, pv, pi);
  k_merge<<<dim3((BTOK+255)/256), 256, 0, stream>>>(pv, pi, cnd);
  k_stage2<<<dim3(BTOK), 128, 0, stream>>>(x, W_enc, K_all, V_all, cnd, out);
}
